// LookaheadModel_35270271435280
// MI455X (gfx1250) — matrix operand order repaired and checked by reading, not compiled, not run on hardware
//
#include <hip/hip_runtime.h>
#include <math.h>

// ---------------- problem constants ----------------
#define Bsz   16
#define Tn    2048
#define Hn    512
#define H2    1024
#define Vn    50257
#define NCn   2045      // T - 3
#define KSEL  512       // FWD_SLOTS
#define NSKIPn 1533     // NC - K
#define NSKPAD 1536
#define RSEL  128       // RETRO_SLOTS actually used
#define MEMS  640       // MEM_SLOTS == KSEL + RSEL
#define WND   8

// ---------------- WMMA types ----------------
typedef __attribute__((ext_vector_type(16))) __bf16 bf16x16;
typedef __attribute__((ext_vector_type(8)))  float  f32x8;

union BFrag { bf16x16 v; int4 q[2]; };

__device__ __forceinline__ unsigned short f32_to_bf16(float f) {
    unsigned u = __float_as_uint(f);
    u = u + 0x7FFFu + ((u >> 16) & 1u);   // round-to-nearest-even
    return (unsigned short)(u >> 16);
}

// A fragment: 16x32 (MxK) bf16. Lane L: M = L&15, half = L>>4.
// elements 0..7  -> K = k0 + half*8 + e
// elements 8..15 -> K = k0 + 16 + half*8 + (e-8)
__device__ __forceinline__ bf16x16 load_a_frag(const unsigned short* base,
                                               int row, int stride, int k0, int hl) {
    BFrag f;
    const unsigned short* p = base + row * stride + k0 + hl * 8;
    f.q[0] = *(const int4*)(p);
    f.q[1] = *(const int4*)(p + 16);
    return f.v;
}

// B fragment: 32x16 (KxN) bf16, source stored transposed (N-major, K contiguous).
// Lane L: N = L&15 (folded into `row`), half = L>>4, elements 0..7 -> K = k0 + half*8 + e, 8..15 -> K = k0 + 16 + half*8 + (e-8)
__device__ __forceinline__ bf16x16 load_b_frag(const unsigned short* base,
                                               long row, int stride, int k0, int hl) {
    BFrag f;
    const unsigned short* p = base + row * (long)stride + k0 + hl * 8;
    f.q[0] = *(const int4*)(p);
    f.q[1] = *(const int4*)(p + 16);
    return f.v;
}

__device__ __forceinline__ f32x8 wmma_bf16(bf16x16 a, bf16x16 b, f32x8 c) {
    return __builtin_amdgcn_wmma_f32_16x16x32_bf16(false, a, false, b, (short)0, c, false, false);
}

// ---------------- K0: transpose + fp32 -> bf16 weight conversion ----------------
// dst[n*rows + k] = bf16(src[k*cols + n]);  dst shape: cols x rows (N-major, K contiguous)
__global__ void wt_transpose_bf16(const float* __restrict__ src,
                                  unsigned short* __restrict__ dst,
                                  int rows, int cols) {
    int i = blockIdx.x * 256 + threadIdx.x;
    if (i >= rows * cols) return;
    int n = i / rows, k = i - n * rows;
    dst[i] = f32_to_bf16(src[(long)k * cols + n]);
}

// ---------------- K1: embed gather + FFN + residual + LayerNorm + fwd scores ----
__global__ __launch_bounds__(256)
void ffn_ln_kernel(const int* __restrict__ seq,
                   const float* __restrict__ embed,
                   const float* __restrict__ ff1_b,
                   const float* __restrict__ ff2_b,
                   const float* __restrict__ ln_w,
                   const float* __restrict__ ln_b,
                   const float* __restrict__ fg_w,
                   const float* __restrict__ fg_b,
                   const unsigned short* __restrict__ ff1T,  // [1024][512] bf16
                   const unsigned short* __restrict__ ff2T,  // [512][1024] bf16
                   float* __restrict__ hidden,               // [B*T][512]
                   float* __restrict__ fwd_scores)           // [B][2048]
{
    __shared__ __align__(16) unsigned char regionA[16 * 512 * 4]; // a0(bf16) then h(f32)
    __shared__ __align__(16) unsigned short actS[16 * 512];       // act half, bf16
    __shared__ int seqS[16];

    const int tid  = threadIdx.x;
    const int w    = tid >> 5;
    const int lane = tid & 31;
    const int nlo  = lane & 15;
    const int hl   = lane >> 4;

    const int g0 = blockIdx.x * 16;            // flat token base (b*T + t0)
    unsigned short* a0S = (unsigned short*)regionA;

    if (tid < 16) seqS[tid] = seq[g0 + tid];
    __syncthreads();

    // gather 16 embedding rows as bf16
    for (int i = tid; i < 16 * 512; i += 256) {
        int r = i >> 9, c = i & 511;
        a0S[i] = f32_to_bf16(embed[(long)seqS[r] * Hn + c]);
    }
    __syncthreads();

    f32x8 acc2[4];

    for (int hn = 0; hn < 2; ++hn) {
        // ---- FF1: act[:, hn*512 .. +512) = relu(a0 @ W1 + b1) ----
        for (int j = 0; j < 4; ++j) {
            int nG = hn * 512 + w * 64 + j * 16 + nlo;   // output col in [0,1024)
            f32x8 acc;
            float bv = ff1_b[nG];
            #pragma unroll
            for (int r = 0; r < 8; ++r) acc[r] = bv;
            for (int k0 = 0; k0 < 512; k0 += 32) {
                bf16x16 af = load_a_frag(a0S, nlo, 512, k0, hl);
                bf16x16 bfv = load_b_frag(ff1T, nG, 512, k0, hl);
                acc = wmma_bf16(af, bfv, acc);
            }
            int cl = w * 64 + j * 16 + nlo;              // local col in act half
            #pragma unroll
            for (int r = 0; r < 8; ++r) {
                float v = acc[r] > 0.f ? acc[r] : 0.f;
                actS[(r + 8 * hl) * 512 + cl] = f32_to_bf16(v);
            }
        }
        __syncthreads();

        // ---- FF2 partial: acc2 += act_half @ W2[hn*512 .. ,:] ----
        for (int j = 0; j < 4; ++j) {
            int n2 = w * 64 + j * 16 + nlo;              // output col in [0,512)
            if (hn == 0) {
                float bv = ff2_b[n2];
                #pragma unroll
                for (int r = 0; r < 8; ++r) acc2[j][r] = bv;
            }
            for (int k0 = 0; k0 < 512; k0 += 32) {
                bf16x16 af = load_a_frag(actS, nlo, 512, k0, hl);
                bf16x16 bfv = load_b_frag(ff2T, n2, 1024, hn * 512 + k0, hl);
                acc2[j] = wmma_bf16(af, bfv, acc2[j]);
            }
        }
        __syncthreads();
    }

    // ---- epilogue: residual add, stage h fp32 in LDS ----
    float* hS = (float*)regionA;  // a0 dead now
    for (int j = 0; j < 4; ++j) {
        int n2 = w * 64 + j * 16 + nlo;
        #pragma unroll
        for (int r = 0; r < 8; ++r) {
            int M = r + 8 * hl;
            float h0v = embed[(long)seqS[M] * Hn + n2];
            hS[M * 512 + n2] = acc2[j][r] + h0v;
        }
    }
    __syncthreads();

    // ---- LayerNorm + fwd gate score (2 rows per wave) ----
    for (int rr = 0; rr < 2; ++rr) {
        int m = w * 2 + rr;
        float s = 0.f, s2 = 0.f;
        for (int i = lane; i < 512; i += 32) {
            float v = hS[m * 512 + i];
            s += v; s2 += v * v;
        }
        #pragma unroll
        for (int off = 16; off > 0; off >>= 1) {
            s  += __shfl_xor(s,  off);
            s2 += __shfl_xor(s2, off);
        }
        float mu  = s * (1.f / 512.f);
        float var = s2 * (1.f / 512.f) - mu * mu;
        float rstd = rsqrtf(var + 1e-5f);

        int gt = g0 + m;
        float sc = 0.f;
        for (int i = lane; i < 512; i += 32) {
            float v = (hS[m * 512 + i] - mu) * rstd * ln_w[i] + ln_b[i];
            hidden[(long)gt * Hn + i] = v;
            sc += v * fg_w[i];
        }
        #pragma unroll
        for (int off = 16; off > 0; off >>= 1) sc += __shfl_xor(sc, off);
        int t = gt & (Tn - 1);
        int b = gt >> 11;  // /2048
        if (lane == 0 && t < NCn) fwd_scores[b * 2048 + t] = sc + fg_b[0];
    }
}

// ---------------- K2: top-K=512 forward selection + skipped list ----------------
__global__ __launch_bounds__(256)
void topk_fwd_kernel(const float* __restrict__ fwd_scores,
                     int* __restrict__ mem_idx,   // [B][640]
                     int* __restrict__ skipped)   // [B][1536]
{
    __shared__ float scS[2048];
    __shared__ unsigned char selS[2048];
    int b = blockIdx.x, tid = threadIdx.x;
    for (int i = tid; i < NCn; i += 256) scS[i] = fwd_scores[b * 2048 + i];
    __syncthreads();
    for (int i = tid; i < NCn; i += 256) {
        float si = scS[i]; int rank = 0;
        for (int j = 0; j < NCn; ++j) {
            float sj = scS[j];
            rank += (sj > si) || (sj == si && j < i);
        }
        selS[i] = (rank < KSEL) ? 1 : 0;
    }
    __syncthreads();
    for (int i = tid; i < NCn; i += 256) {
        int nsel = 0;
        for (int j = 0; j < i; ++j) nsel += selS[j];
        if (selS[i]) mem_idx[b * MEMS + nsel] = i;
        else         skipped[b * NSKPAD + (i - nsel)] = i;
    }
}

// ---------------- K3: retro gate MLP (WMMA) --------------------------------
__global__ __launch_bounds__(256)
void retro_gate_kernel(const float* __restrict__ hidden,
                       const int* __restrict__ skipped,
                       const unsigned short* __restrict__ rg1T, // [512][1024] bf16
                       const float* __restrict__ rg1_b,
                       const float* __restrict__ rg2_w,
                       const float* __restrict__ rg2_b,
                       float* __restrict__ retro_sc)            // [B][1536]
{
    __shared__ __align__(16) unsigned short giS[16 * 1024]; // 32KB bf16
    __shared__ float redS[8][32][8];
    __shared__ int skS[16];

    const int tid  = threadIdx.x;
    const int w    = tid >> 5;
    const int lane = tid & 31;
    const int nlo  = lane & 15;
    const int hl   = lane >> 4;
    const int b    = blockIdx.y;
    const int tile = blockIdx.x;

    if (tid < 16) {
        int si = tile * 16 + tid;
        skS[tid] = (si < NSKIPn) ? skipped[b * NSKPAD + si] : 0;
    }
    __syncthreads();

    const float* hb = hidden + (long)b * Tn * Hn;
    // build gi = [hid[s], windowed-ctx] in bf16
    for (int i = tid; i < 16 * 1024; i += 256) {
        int r = i >> 10, c = i & 1023;
        int s = skS[r];
        float v;
        if (c < 512) {
            v = hb[(long)s * Hn + c];
        } else {
            int cc = c - 512;
            int tend = s + 1 + WND; if (tend > NCn) tend = NCn;
            int cnt = tend - s - 1;
            if (cnt > 0) {
                float a = 0.f;
                for (int u = s + 1; u < tend; ++u) a += hb[(long)u * Hn + cc];
                v = a / (float)cnt;
            } else {
                v = hb[(long)s * Hn + cc];
            }
        }
        giS[i] = f32_to_bf16(v);
    }
    __syncthreads();

    float part[8];
    #pragma unroll
    for (int r = 0; r < 8; ++r) part[r] = 0.f;

    for (int j = 0; j < 4; ++j) {
        int n = w * 64 + j * 16 + nlo;
        f32x8 acc;
        float bv = rg1_b[n];
        #pragma unroll
        for (int r = 0; r < 8; ++r) acc[r] = bv;
        for (int k0 = 0; k0 < 1024; k0 += 32) {
            bf16x16 af  = load_a_frag(giS, nlo, 1024, k0, hl);
            bf16x16 bfv = load_b_frag(rg1T, n, 1024, k0, hl);
            acc = wmma_bf16(af, bfv, acc);
        }
        float w2 = rg2_w[n];
        #pragma unroll
        for (int r = 0; r < 8; ++r) part[r] += fmaxf(acc[r], 0.f) * w2;
    }
    #pragma unroll
    for (int r = 0; r < 8; ++r) redS[w][lane][r] = part[r];
    __syncthreads();

    if (tid < 16) {
        int m = tid;
        float tot = rg2_b[0];
        for (int ww = 0; ww < 8; ++ww)
            for (int l2 = 0; l2 < 16; ++l2)
                tot += redS[ww][(m >> 3) * 16 + l2][m & 7];
        float g = 1.f / (1.f + expf(-tot));
        int si = tile * 16 + m;
        if (si < NSKIPn) retro_sc[b * NSKPAD + si] = g;
    }
}

// ---------------- K4: top-R=128 retro selection (rank order = score order) ----
__global__ __launch_bounds__(256)
void topk_retro_kernel(const float* __restrict__ retro_sc,
                       const int* __restrict__ skipped,
                       int* __restrict__ mem_idx)
{
    __shared__ float scS[NSKPAD];
    int b = blockIdx.x, tid = threadIdx.x;
    for (int i = tid; i < NSKIPn; i += 256) scS[i] = retro_sc[b * NSKPAD + i];
    __syncthreads();
    for (int i = tid; i < NSKIPn; i += 256) {
        float si = scS[i]; int rank = 0;
        for (int j = 0; j < NSKIPn; ++j) {
            float sj = scS[j];
            rank += (sj > si) || (sj == si && j < i);
        }
        if (rank < RSEL) mem_idx[b * MEMS + KSEL + rank] = skipped[b * NSKPAD + i];
    }
}

// ---------------- K5: query proj + attention over 640 memory slots ------------
__global__ __launch_bounds__(256)
void attn_kernel(const float* __restrict__ hidden,
                 const int* __restrict__ mem_idx,
                 const float* __restrict__ q_w,
                 const float* __restrict__ q_b,
                 float* __restrict__ ctx)       // [B][512]
{
    __shared__ float hq[512];
    __shared__ float qS[512];
    __shared__ float scS[MEMS];
    __shared__ int idxS[MEMS];
    __shared__ float red[256];

    int b = blockIdx.x, tid = threadIdx.x;
    const float* hb = hidden + (long)b * Tn * Hn;

    for (int i = tid; i < 512; i += 256) hq[i] = hb[(long)(Tn - 2) * Hn + i];
    for (int i = tid; i < MEMS; i += 256) idxS[i] = mem_idx[b * MEMS + i];
    __syncthreads();

    for (int j = tid; j < 512; j += 256) {
        float a = q_b[j];
        for (int i = 0; i < 512; ++i) a += hq[i] * q_w[(long)i * Hn + j];
        qS[j] = a;
    }
    __syncthreads();

    for (int m = tid; m < MEMS; m += 256) {
        const float* row = hb + (long)idxS[m] * Hn;
        float a = 0.f;
        for (int i = 0; i < 512; ++i) a += row[i] * qS[i];
        scS[m] = a;   // mask all ones: K + R == MEM_SLOTS
    }
    __syncthreads();

    // softmax
    float lm = -1e30f;
    for (int m = tid; m < MEMS; m += 256) lm = fmaxf(lm, scS[m]);
    red[tid] = lm; __syncthreads();
    for (int s = 128; s > 0; s >>= 1) { if (tid < s) red[tid] = fmaxf(red[tid], red[tid + s]); __syncthreads(); }
    float mmax = red[0]; __syncthreads();
    float ls = 0.f;
    for (int m = tid; m < MEMS; m += 256) { float e = expf(scS[m] - mmax); scS[m] = e; ls += e; }
    red[tid] = ls; __syncthreads();
    for (int s = 128; s > 0; s >>= 1) { if (tid < s) red[tid] += red[tid + s]; __syncthreads(); }
    float rsum = 1.f / red[0]; __syncthreads();

    for (int h = tid; h < 512; h += 256) {
        float c = 0.f;
        for (int m = 0; m < MEMS; ++m) c += scS[m] * hb[(long)idxS[m] * Hn + h];
        ctx[b * Hn + h] = c * rsum;
    }
}

// ---------------- K6: vocab projection (HBM-bound streaming) ------------------
__global__ __launch_bounds__(256)
void vocab_proj_kernel(const float* __restrict__ ctx,
                       const float* __restrict__ out_w,
                       const float* __restrict__ out_b,
                       float* __restrict__ out)
{
    __shared__ float ctxS[16 * 512];
    int tid = threadIdx.x;
    for (int i = tid; i < 16 * 512; i += 256) ctxS[i] = ctx[i];
    __syncthreads();

    int v = blockIdx.x * 256 + tid;
    if (v >= Vn) return;
    float acc[16];
    #pragma unroll
    for (int b = 0; b < 16; ++b) acc[b] = 0.f;
    for (int h = 0; h < 512; ++h) {
        float wv = out_w[(long)h * Vn + v];
        #pragma unroll
        for (int b = 0; b < 16; ++b) acc[b] += ctxS[b * 512 + h] * wv;
    }
    float bb = out_b[v];
    #pragma unroll
    for (int b = 0; b < 16; ++b) out[(long)b * Vn + v] = acc[b] + bb;
}

// ---------------- launch ------------------------------------------------------
extern "C" void kernel_launch(void* const* d_in, const int* in_sizes, int n_in,
                              void* d_out, int out_size, void* d_ws, size_t ws_size,
                              hipStream_t stream) {
    const int*   seq    = (const int*)  d_in[0];
    const float* embed  = (const float*)d_in[1];
    const float* ff1_w  = (const float*)d_in[2];
    const float* ff1_b  = (const float*)d_in[3];
    const float* ff2_w  = (const float*)d_in[4];
    const float* ff2_b  = (const float*)d_in[5];
    const float* ln_w   = (const float*)d_in[6];
    const float* ln_b   = (const float*)d_in[7];
    const float* fg_w   = (const float*)d_in[8];
    const float* fg_b   = (const float*)d_in[9];
    const float* rg1_w  = (const float*)d_in[10];
    const float* rg1_b  = (const float*)d_in[11];
    const float* rg2_w  = (const float*)d_in[12];
    const float* rg2_b  = (const float*)d_in[13];
    const float* q_w    = (const float*)d_in[14];
    const float* q_b    = (const float*)d_in[15];
    const float* out_w  = (const float*)d_in[16];
    const float* out_b  = (const float*)d_in[17];
    float* out = (float*)d_out;

    char* ws = (char*)d_ws;
    size_t off = 0;
    float* hidden = (float*)(ws + off);            off += (size_t)Bsz * Tn * Hn * 4;   // 64 MB
    unsigned short* ff1T = (unsigned short*)(ws + off); off += (size_t)H2 * Hn * 2;    // 1 MB
    unsigned short* ff2T = (unsigned short*)(ws + off); off += (size_t)Hn * H2 * 2;    // 1 MB
    unsigned short* rg1T = (unsigned short*)(ws + off); off += (size_t)Hn * H2 * 2;    // 1 MB
    float* fsc    = (float*)(ws + off);            off += (size_t)Bsz * 2048 * 4;
    int*   memidx = (int*)  (ws + off);            off += (size_t)Bsz * MEMS * 4;
    int*   skip   = (int*)  (ws + off);            off += (size_t)Bsz * NSKPAD * 4;
    float* rsc    = (float*)(ws + off);            off += (size_t)Bsz * NSKPAD * 4;
    float* ctx    = (float*)(ws + off);            off += (size_t)Bsz * Hn * 4;

    // K0: weight transpose+convert (ff1: 512x1024, ff2: 1024x512, rg1: 1024x512)
    wt_transpose_bf16<<<(Hn * H2 + 255) / 256, 256, 0, stream>>>(ff1_w, ff1T, Hn, H2);
    wt_transpose_bf16<<<(H2 * Hn + 255) / 256, 256, 0, stream>>>(ff2_w, ff2T, H2, Hn);
    wt_transpose_bf16<<<(H2 * Hn + 255) / 256, 256, 0, stream>>>(rg1_w, rg1T, H2, Hn);

    // K1: embed + FFN + LN + fwd scores
    ffn_ln_kernel<<<(Bsz * Tn) / 16, 256, 0, stream>>>(
        seq, embed, ff1_b, ff2_b, ln_w, ln_b, fg_w, fg_b, ff1T, ff2T, hidden, fsc);

    // K2: forward top-K + skipped
    topk_fwd_kernel<<<Bsz, 256, 0, stream>>>(fsc, memidx, skip);

    // K3: retro gate MLP
    retro_gate_kernel<<<dim3(NSKPAD / 16, Bsz), 256, 0, stream>>>(
        hidden, skip, rg1T, rg1_b, rg2_w, rg2_b, rsc);

    // K4: retro top-R
    topk_retro_kernel<<<Bsz, 256, 0, stream>>>(rsc, skip, memidx);

    // K5: query + attention
    attn_kernel<<<Bsz, 256, 0, stream>>>(hidden, memidx, q_w, q_b, ctx);

    // K6: vocab projection
    vocab_proj_kernel<<<(Vn + 255) / 256, 256, 0, stream>>>(ctx, out_w, out_b, out);
}
